// Spatial_RNN_10574209483241
// MI455X (gfx1250) — matrix operand order repaired and checked by reading, not compiled, not run on hardware
//
#include <hip/hip_runtime.h>

typedef __attribute__((ext_vector_type(16))) __bf16 v16bf;
typedef __attribute__((ext_vector_type(8)))  float  v8f;

#define PIXT 4   // pixel sub-tiles (16 px each) per wave -> 16co x 64px tile

// ------------------------------------------------------------------
// Weight prepack: fp32 OIHW -> bf16 [Cout16][Kpad], zero padded
// ------------------------------------------------------------------
__global__ void prepack_w(const float* __restrict__ w, __bf16* __restrict__ wbf,
                          int Cout, int K, int Kpad, int Cout16) {
  int total = Cout16 * Kpad;
  for (int i = blockIdx.x * blockDim.x + threadIdx.x; i < total;
       i += gridDim.x * blockDim.x) {
    int co = i / Kpad, k = i - co * Kpad;
    float v = (co < Cout && k < K) ? w[co * K + k] : 0.0f;
    wbf[i] = (__bf16)v;
  }
}

// ------------------------------------------------------------------
// Implicit-GEMM conv (same-size, stride 1) via v_wmma_f32_16x16x32_bf16
// One wave computes a 16(outch) x 64(pixel) tile: A loaded once per
// K-step (8 dword loads), 4 WMMAs issued per K-step.
// A fragment (16x32 bf16, contiguous k):
//   lane<16: M=lane, VGPR v holds K = v*2 (+1)
//   lane>=16: same + 16
// B fragment (32x16 bf16):
//   lane<16: N=lane, K = kb+0..15 ; lane>=16: N=lane-16, K = kb+16..31
// C/D f32: VGPR v -> M = v + (lane>=16 ? 8:0), N = lane&15
// ------------------------------------------------------------------
__global__ void conv_wmma(const float* __restrict__ x, const __bf16* __restrict__ wbf,
                          const float* __restrict__ bias, float* __restrict__ out,
                          int Cin, int Cout, int H, int W,
                          int KH, int KW, int pad, int Kreal, int Kpad,
                          int nCo, int nPix, int relu, int totalWaves) {
  int wave = blockIdx.x * (blockDim.x >> 5) + (threadIdx.x >> 5);
  if (wave >= totalWaves) return;          // whole-wave exit: EXEC stays full
  int lane = threadIdx.x & 31;
  int P = H * W;
  int tilesPerImg = nCo * nPix;
  int n   = wave / tilesPerImg;
  int rem = wave - n * tilesPerImg;
  int coT = rem / nPix;
  int coBase  = coT * 16;
  int pixBase = (rem - coT * nPix) * (16 * PIXT);

  int  l15 = lane & 15;
  bool hi  = lane >= 16;

  // per-subtile pixel coords (clamped; OOB lanes compute garbage, store guarded)
  int oh[PIXT], ow[PIXT];
#pragma unroll
  for (int t = 0; t < PIXT; ++t) {
    int p = pixBase + 16 * t + l15;
    if (p >= P) p = P - 1;
    oh[t] = p / W;
    ow[t] = p - oh[t] * W;
  }
  const float* xn = x + (long long)n * Cin * P;

  int ksz = KH * KW;
  const unsigned* arow32 =
      (const unsigned*)(wbf + (long long)(coBase + l15) * Kpad);
  int abase = hi ? 16 : 0;

  v8f acc[PIXT] = {};
  for (int kb = 0; kb < Kpad; kb += 32) {
    // ---- A fragment: 8 aligned dword loads (bf16 pairs) ----
    union { unsigned u[8]; v16bf v; } au;
#pragma unroll
    for (int v = 0; v < 8; ++v) {
      int pairBase = (kb + abase) >> 1;
      au.u[v] = arow32[pairBase + v];
    }
    // ---- speculative prefetch of next K-step's input row ----
    if (kb + 32 < Kpad) {
      int k2  = kb + 32 + (hi ? 16 : 0);
      int ci2 = k2 / ksz; if (ci2 >= Cin) ci2 = Cin - 1;
      __builtin_prefetch(&xn[(ci2 * H + oh[0]) * W + ow[0]], 0, 1);
    }
    // ---- B fragments: on-the-fly im2col gather, 4 sub-tiles ----
    int k  = kb + (hi ? 16 : 0);
    int ci = k / ksz;
    int rs = k - ci * ksz;
    int r  = rs / KW;
    int s  = rs - r * KW;
    v16bf bf[PIXT];
#pragma unroll
    for (int j = 0; j < 16; ++j) {
      bool kin = (k + j < Kreal);
      int rr = r - pad;
      int ss = s - pad;
      int rowBase = ci * H;
#pragma unroll
      for (int t = 0; t < PIXT; ++t) {
        float val = 0.0f;
        if (kin) {
          int ih = oh[t] + rr;
          int iw = ow[t] + ss;
          if ((unsigned)ih < (unsigned)H && (unsigned)iw < (unsigned)W)
            val = xn[(rowBase + ih) * W + iw];
        }
        bf[t][j] = (__bf16)val;
      }
      if (++s == KW) { s = 0; if (++r == KH) { r = 0; ++ci; } }
    }
#pragma unroll
    for (int t = 0; t < PIXT; ++t)
      acc[t] = __builtin_amdgcn_wmma_f32_16x16x32_bf16(false, au.v, false, bf[t],
                                                       (short)0, acc[t], false, false);
  }

  // ---- epilogue: bias + optional ReLU, guarded stores ----
  int coR = coBase + (hi ? 8 : 0);
  float bv[8];
#pragma unroll
  for (int v = 0; v < 8; ++v)
    bv[v] = (coR + v < Cout) ? bias[coR + v] : 0.0f;
  float* on = out + (long long)n * Cout * P;
#pragma unroll
  for (int t = 0; t < PIXT; ++t) {
    int pixN = pixBase + 16 * t + l15;
    if (pixN < P) {
#pragma unroll
      for (int v = 0; v < 8; ++v) {
        int co = coR + v;
        if (co < Cout) {
          float rv = acc[t][v] + bv[v];
          if (relu) rv = fmaxf(rv, 0.0f);
          on[co * P + pixN] = rv;
        }
      }
    }
  }
}

// ------------------------------------------------------------------
// 2x2 max pool
// ------------------------------------------------------------------
__global__ void maxpool2_k(const float* __restrict__ in, float* __restrict__ out,
                           int NC, int H, int W) {
  int Ho = H >> 1, Wo = W >> 1;
  long long total = (long long)NC * Ho * Wo;
  for (long long i = (long long)blockIdx.x * blockDim.x + threadIdx.x; i < total;
       i += (long long)gridDim.x * blockDim.x) {
    int wo = (int)(i % Wo);
    long long t = i / Wo;
    int ho = (int)(t % Ho);
    int nc = (int)(t / Ho);
    const float* p = in + ((long long)nc * H + 2 * ho) * W + 2 * wo;
    out[i] = fmaxf(fmaxf(p[0], p[1]), fmaxf(p[W], p[W + 1]));
  }
}

// ------------------------------------------------------------------
// 2x bilinear upsample (half-pixel centers, edge clamp) + channel concat
// out channels [0,Cu): up2(up) ; [Cu,Cu+Cs): copy of skip
// ------------------------------------------------------------------
__global__ void upcat_k(const float* __restrict__ up, const float* __restrict__ skip,
                        float* __restrict__ out, int NB, int Cu, int Cs, int h, int w) {
  int H2 = 2 * h, W2 = 2 * w, Ct = Cu + Cs;
  long long total = (long long)NB * Ct * H2 * W2;
  for (long long idx = (long long)blockIdx.x * blockDim.x + threadIdx.x; idx < total;
       idx += (long long)gridDim.x * blockDim.x) {
    int j = (int)(idx % W2);
    long long t = idx / W2;
    int i = (int)(t % H2); t /= H2;
    int c = (int)(t % Ct);
    int n = (int)(t / Ct);
    float val;
    if (c < Cu) {
      float fy = 0.5f * i - 0.25f;
      float fx = 0.5f * j - 0.25f;
      int y0 = (int)floorf(fy); float wy = fy - (float)y0;
      int x0 = (int)floorf(fx); float wx = fx - (float)x0;
      int y0c = min(max(y0, 0), h - 1), y1c = min(max(y0 + 1, 0), h - 1);
      int x0c = min(max(x0, 0), w - 1), x1c = min(max(x0 + 1, 0), w - 1);
      const float* s = up + ((long long)n * Cu + c) * h * w;
      float v00 = s[y0c * w + x0c], v01 = s[y0c * w + x1c];
      float v10 = s[y1c * w + x0c], v11 = s[y1c * w + x1c];
      val = (1.0f - wy) * ((1.0f - wx) * v00 + wx * v01)
          +         wy  * ((1.0f - wx) * v10 + wx * v11);
    } else {
      val = skip[((long long)n * Cs + (c - Cu)) * (long long)H2 * W2 + (long long)i * W2 + j];
    }
    out[idx] = val;
  }
}

// ------------------------------------------------------------------
// LRNN: h_t = p_t*h_{t-1} + (1-p_t)*x_t, scanned over the fused (b,n) axis
// exactly like lax.scan in the reference (carry crosses batch images).
// Chunked linear scan: chunk = one image (96 steps), 128 chunks.
// Thread mapping for phases 1/3: tid -> (d, c, b, m), m fastest.
// Direction physical coords: d0 l2r (h=m,w=n)  d1 r2l (h=m,w=95-n)
//                            d2 t2b (h=n,w=m)  d3 b2t (h=95-n,w=m)
// ------------------------------------------------------------------
__device__ __forceinline__ void lrnn_coords(int d, int n, int m, int& hh, int& ww) {
  if (d == 0)      { hh = m;      ww = n;      }
  else if (d == 1) { hh = m;      ww = 95 - n; }
  else if (d == 2) { hh = n;      ww = m;      }
  else             { hh = 95 - n; ww = m;      }
}

__global__ void lrnn_phase1(const float* __restrict__ X, const float* __restrict__ FM,
                            float* __restrict__ Afac, float* __restrict__ Bfac) {
  int tid = blockIdx.x * blockDim.x + threadIdx.x;
  if (tid >= 4 * 16 * 128 * 96) return;
  int m = tid % 96;
  int t = tid / 96;
  int b = t % 128; t /= 128;
  int c = t % 16;
  int d = t / 16;
  const float* xb = X  + ((long long)b * 16 + c) * 9216;
  const float* pb = FM + ((long long)b * 64 + d * 16 + c) * 9216;
  float a = 1.0f, h = 0.0f;
  for (int n = 0; n < 96; ++n) {
    int hh, ww; lrnn_coords(d, n, m, hh, ww);
    float xv = xb[hh * 96 + ww];
    float pv = pb[hh * 96 + ww];
    a *= pv;
    h = (1.0f - pv) * xv + pv * h;
  }
  Afac[tid] = a;   // composition: h_out = a*h_in + b
  Bfac[tid] = h;
}

__global__ void lrnn_phase2(const float* __restrict__ Afac, const float* __restrict__ Bfac,
                            float* __restrict__ H0) {
  int tid = blockIdx.x * blockDim.x + threadIdx.x;
  if (tid >= 4 * 16 * 96) return;
  int m = tid % 96;
  int t = tid / 96;
  int c = t % 16;
  int d = t / 16;
  float carry = 0.0f;                       // scan init = zeros
  for (int b = 0; b < 128; ++b) {
    long long idx = (((long long)d * 16 + c) * 128 + b) * 96 + m;
    H0[idx] = carry;
    carry = Afac[idx] * carry + Bfac[idx];
  }
}

__global__ void lrnn_phase3(const float* __restrict__ X, const float* __restrict__ FM,
                            const float* __restrict__ H0,
                            float* __restrict__ hbA, float* __restrict__ hb3,
                            long long strideD) {
  int tid = blockIdx.x * blockDim.x + threadIdx.x;
  if (tid >= 4 * 16 * 128 * 96) return;
  int m = tid % 96;
  int t = tid / 96;
  int b = t % 128; t /= 128;
  int c = t % 16;
  int d = t / 16;
  const float* xb = X  + ((long long)b * 16 + c) * 9216;
  const float* pb = FM + ((long long)b * 64 + d * 16 + c) * 9216;
  float h = H0[tid];
  float* hb = (d < 3) ? (hbA + (long long)d * strideD) : hb3;
  float* ob = hb + ((long long)b * 16 + c) * 9216;   // out[b][c][scan_n][m]
  for (int n = 0; n < 96; ++n) {
    int hh, ww; lrnn_coords(d, n, m, hh, ww);
    float xv = xb[hh * 96 + ww];
    float pv = pb[hh * 96 + ww];
    h = (1.0f - pv) * xv + pv * h;
    ob[n * 96 + m] = h;
  }
}

__global__ void merge_k(const float* __restrict__ hbA, const float* __restrict__ hb3,
                        float* __restrict__ out, long long strideD, long long total) {
  for (long long i = (long long)blockIdx.x * blockDim.x + threadIdx.x; i < total;
       i += (long long)gridDim.x * blockDim.x) {
    float v = fmaxf(fmaxf(hbA[i], hbA[i + strideD]),
                    fmaxf(hbA[i + 2 * strideD], hb3[i]));
    out[i] = v;
  }
}

// ------------------------------------------------------------------
// Host orchestration
// ------------------------------------------------------------------
extern "C" void kernel_launch(void* const* d_in, const int* in_sizes, int n_in,
                              void* d_out, int out_size, void* d_ws, size_t ws_size,
                              hipStream_t stream) {
  const float* img = (const float*)d_in[0];
  const float* c1w = (const float*)d_in[1];  const float* c1b = (const float*)d_in[2];
  const float* c2w = (const float*)d_in[3];  const float* c2b = (const float*)d_in[4];
  const float* c3w = (const float*)d_in[5];  const float* c3b = (const float*)d_in[6];
  const float* c4w = (const float*)d_in[7];  const float* c4b = (const float*)d_in[8];
  const float* c5w = (const float*)d_in[9];  const float* c5b = (const float*)d_in[10];
  const float* c6w = (const float*)d_in[11]; const float* c6b = (const float*)d_in[12];
  const float* c7w = (const float*)d_in[13]; const float* c7b = (const float*)d_in[14];
  const float* c8w = (const float*)d_in[15]; const float* c8b = (const float*)d_in[16];
  const float* c9w = (const float*)d_in[17]; const float* c9b = (const float*)d_in[18];
  const float* inw = (const float*)d_in[19]; const float* inb = (const float*)d_in[20];
  const float* outw= (const float*)d_in[21]; const float* outb= (const float*)d_in[22];

  char* ws = (char*)d_ws;
  size_t off = 0;
  auto allocF = [&](size_t nf) {
    float* p = (float*)(ws + off);
    off += ((nf * sizeof(float) + 255) & ~(size_t)255);
    return p;
  };
  auto allocBF = [&](size_t nb) {
    __bf16* p = (__bf16*)(ws + off);
    off += ((nb * sizeof(__bf16) + 255) & ~(size_t)255);
    return p;
  };

  const int NB = 128;
  // activation buffers (floats)
  float* fm      = allocF(75497472);  // 128*64*96*96
  float* lrnnin  = allocF(18874368);  // 128*16*96*96
  float* x1      = allocF(18874368);  // 128*16*96*96  (reused as `merged` at the end)
  float* pl1     = allocF(4718592);   // 128*16*48*48
  float* x2      = allocF(9437184);   // 128*32*48*48
  float* pl2     = allocF(2359296);   // 128*32*24*24
  float* x3      = allocF(2359296);
  float* pl3     = allocF(589824);    // 128*32*12*12
  float* x4      = allocF(589824);
  float* pl4     = allocF(147456);    // 128*32*6*6
  float* x5      = allocF(294912);    // 128*64*6*6
  float* cat6    = allocF(1769472);   // 128*96*12*12
  float* x6      = allocF(589824);
  float* cat7    = allocF(4718592);   // 128*64*24*24
  float* x7      = allocF(2359296);
  float* cat8    = allocF(18874368);  // 128*64*48*48 (reused as hbuf[3])
  float* x8      = allocF(9437184);   // 128*32*48*48
  float* cat9    = allocF(56623104);  // 128*48*96*96 (reused as hbuf[0..2])
  float* Afac    = allocF(786432);    // 4*16*128*96
  float* Bfac    = allocF(786432);
  float* H0      = allocF(786432);

  // prepack all conv weights to bf16 GEMM layout
  auto prep = [&](const float* w, int Cout, int Cin, int KH, int KW) -> __bf16* {
    int K = Cin * KH * KW, Kpad = (K + 31) & ~31, Cout16 = (Cout + 15) & ~15;
    __bf16* wb = allocBF((size_t)Cout16 * Kpad);
    int total = Cout16 * Kpad;
    prepack_w<<<(total + 255) / 256, 256, 0, stream>>>(w, wb, Cout, K, Kpad, Cout16);
    return wb;
  };
  __bf16* wc1 = prep(c1w, 16, 15, 5, 5);
  __bf16* wc2 = prep(c2w, 32, 16, 3, 3);
  __bf16* wc3 = prep(c3w, 32, 32, 3, 3);
  __bf16* wc4 = prep(c4w, 32, 32, 3, 3);
  __bf16* wc5 = prep(c5w, 64, 32, 3, 3);
  __bf16* wc6 = prep(c6w, 32, 96, 3, 3);
  __bf16* wc7 = prep(c7w, 32, 64, 3, 3);
  __bf16* wc8 = prep(c8w, 32, 64, 3, 3);
  __bf16* wc9 = prep(c9w, 64, 48, 3, 3);
  __bf16* win = prep(inw, 16, 15, 3, 3);
  __bf16* wout= prep(outw, 3, 16, 3, 3);

  auto conv = [&](const float* xin, const __bf16* wbf, const float* bias, float* out,
                  int Cin, int Cout, int H, int W, int KH, int KW, int pad, int relu) {
    int Kreal = Cin * KH * KW;
    int Kpad  = (Kreal + 31) & ~31;
    int P = H * W;
    int nCo  = ((Cout + 15) / 16);
    int nPix = (P + 16 * PIXT - 1) / (16 * PIXT);
    int totalWaves = NB * nCo * nPix;
    int blocks = (totalWaves + 7) / 8;       // 8 waves (256 threads) per block
    conv_wmma<<<blocks, 256, 0, stream>>>(xin, wbf, bias, out, Cin, Cout, H, W,
                                          KH, KW, pad, Kreal, Kpad, nCo, nPix,
                                          relu, totalWaves);
  };
  auto pool = [&](const float* in, float* out, int C, int H) {
    long long total = (long long)NB * C * (H / 2) * (H / 2);
    maxpool2_k<<<(int)((total + 255) / 256), 256, 0, stream>>>(in, out, NB * C, H, H);
  };
  auto upcat = [&](const float* up, const float* skip, float* out, int Cu, int Cs, int h) {
    long long total = (long long)NB * (Cu + Cs) * 4 * h * h;
    upcat_k<<<(int)((total + 255) / 256), 256, 0, stream>>>(up, skip, out, NB, Cu, Cs, h, h);
  };

  // ---- encoder ----
  conv(img, wc1, c1b, x1, 15, 16, 96, 96, 5, 5, 2, 1);
  pool(x1, pl1, 16, 96);
  conv(pl1, wc2, c2b, x2, 16, 32, 48, 48, 3, 3, 1, 1);
  pool(x2, pl2, 32, 48);
  conv(pl2, wc3, c3b, x3, 32, 32, 24, 24, 3, 3, 1, 1);
  pool(x3, pl3, 32, 24);
  conv(pl3, wc4, c4b, x4, 32, 32, 12, 12, 3, 3, 1, 1);
  pool(x4, pl4, 32, 12);
  conv(pl4, wc5, c5b, x5, 32, 64, 6, 6, 3, 3, 1, 1);
  // ---- decoder ----
  upcat(x5, x4, cat6, 64, 32, 6);
  conv(cat6, wc6, c6b, x6, 96, 32, 12, 12, 3, 3, 1, 1);
  upcat(x6, x3, cat7, 32, 32, 12);
  conv(cat7, wc7, c7b, x7, 64, 32, 24, 24, 3, 3, 1, 1);
  upcat(x7, x2, cat8, 32, 32, 24);
  conv(cat8, wc8, c8b, x8, 64, 32, 48, 48, 3, 3, 1, 1);
  upcat(x8, x1, cat9, 32, 16, 48);
  conv(cat9, wc9, c9b, fm, 48, 64, 96, 96, 3, 3, 1, 1);
  // ---- LRNN input ----
  conv(img, win, inb, lrnnin, 15, 16, 96, 96, 3, 3, 1, 0);

  // ---- 4-direction linear RNN, chunked parallel scan ----
  const long long strideD = 18874368LL;     // one direction buffer (floats)
  float* hbA = cat9;                        // hbuf[0..2] (cat9 region == 3x strideD)
  float* hb3 = cat8;                        // hbuf[3]
  lrnn_phase1<<<3072, 256, 0, stream>>>(lrnnin, fm, Afac, Bfac);
  lrnn_phase2<<<24, 256, 0, stream>>>(Afac, Bfac, H0);
  lrnn_phase3<<<3072, 256, 0, stream>>>(lrnnin, fm, H0, hbA, hb3, strideD);

  float* merged = x1;                       // x1 region free now
  merge_k<<<(int)((strideD + 255) / 256), 256, 0, stream>>>(hbA, hb3, merged,
                                                            strideD, strideD);
  // ---- output conv ----
  conv(merged, wout, outb, (float*)d_out, 16, 3, 96, 96, 3, 3, 1, 0);
}
